// ALIF_60868276519401
// MI455X (gfx1250) — matrix operand order repaired and checked by reading, not compiled, not run on hardware
//
#include <hip/hip_runtime.h>
#include <hip/hip_bf16.h>

// ---------------------------------------------------------------------------
// Recurrent net step on MI455X (gfx1250, wave32, WMMA).
//   r = sigmoid(3u-2); drive = r @ Wdale + 0.5 + ext[t]; u' = u(1-dt)+drive*dt
// GEMM per step: M=32 (batch), K=2048, N=2048 -> v_wmma_f32_16x16x32_f16.
// Wdale folded once to f16 (8 MB, L2-resident). K split over 4 waves/block,
// LDS reduction, fused epilogue writes usave[t] and r_{t+1}.
// ---------------------------------------------------------------------------

typedef __attribute__((ext_vector_type(16))) _Float16 v16h;
typedef __attribute__((ext_vector_type(8)))  float    v8f;

#define N_    2048
#define B_    32
#define T_    500
#define NOUT_ 2
#define KB_   64    // K blocks of 32 (2048/32)
#define NT_   128   // column tiles of 16 (2048/16)
#define MUE_  0.5f

union Frag16 { v16h v; uint4 q[2]; };

__device__ __forceinline__ float sigmoid_act(float u) {
    return 1.0f / (1.0f + __expf(2.0f - 3.0f * u));
}

// Fold mask * relu(Wrec) * sign (transposed) into packed f16 B-fragments.
// Layout: element ((nt*64 + kb)*32 + lane)*16 + j holds
//   Wdale_T[pre = kb*32 + (j>>3)*16 + (lane>>4)*8 + (j&7)][post = nt*16 + (lane&15)]
// so each lane's 16 halves for one (kb, nt) fragment are contiguous (32 B).
__global__ void prep_w_kernel(const float* __restrict__ wrec,
                              const float* __restrict__ mask,
                              const float* __restrict__ sign,
                              _Float16* __restrict__ wb) {
    int idx = blockIdx.x * blockDim.x + threadIdx.x;
    if (idx >= N_ * N_) return;
    int j    = idx & 15;
    int lane = (idx >> 4) & 31;
    int kb   = (idx >> 9) & 63;
    int nt   = idx >> 15;
    int pre  = kb * 32 + ((j >> 3) << 4) + ((lane >> 4) << 3) + (j & 7);
    int post = nt * 16 + (lane & 15);
    float w = wrec[post * N_ + pre];
    w = w > 0.f ? w : 0.f;
    wb[idx] = (_Float16)(mask[post * N_ + pre] * w * sign[pre]);
}

__global__ void init_r_kernel(const float* __restrict__ u0,
                              _Float16* __restrict__ rbuf) {
    int idx = blockIdx.x * blockDim.x + threadIdx.x;
    if (idx >= B_ * N_) return;
    rbuf[idx] = (_Float16)sigmoid_act(u0[idx]);
}

// out[t][b][o] = sum_k r[b][k] * wout[k][o]   (tiny: 64 dot products of 2048)
__global__ void readout_kernel(const _Float16* __restrict__ rbuf,
                               const float* __restrict__ wout,
                               float* __restrict__ outsave, int t) {
    __shared__ float red[256];
    int b = blockIdx.x >> 1, o = blockIdx.x & 1;
    float s = 0.f;
    for (int k = threadIdx.x; k < N_; k += 256)
        s += (float)rbuf[b * N_ + k] * wout[k * NOUT_ + o];
    red[threadIdx.x] = s;
    __syncthreads();
    for (int off = 128; off > 0; off >>= 1) {
        if ((int)threadIdx.x < off) red[threadIdx.x] += red[threadIdx.x + off];
        __syncthreads();
    }
    if (threadIdx.x == 0)
        outsave[(size_t)t * B_ * NOUT_ + b * NOUT_ + o] = red[0];
}

// One block per 16-column tile. 4 waves split K (16 k-blocks each), partials
// reduced through LDS; waves 0/1 run the fused epilogue for row tiles 0/1.
__global__ __launch_bounds__(128) void step_gemm_kernel(
        const _Float16* __restrict__ rbuf,   // r_t   [32][2048] f16 row-major
        const _Float16* __restrict__ wb,     // packed B fragments
        const float*    __restrict__ uPrev,  // u_t   [32][2048]
        const float*    __restrict__ ext,    // [T][32][2048]
        const float*    __restrict__ dtv,    // dt_tausinv [2048]
        float*          __restrict__ usave,  // [T][32][2048] (output)
        _Float16*       __restrict__ rnext,  // r_{t+1} (ping-pong)
        int t) {
    __shared__ float part[4 * 2 * 32 * 8];   // [wave][frag][lane][8]
    const int lane  = threadIdx.x & 31;
    const int wave  = threadIdx.x >> 5;
    const int nt    = blockIdx.x;
    const int khalf = lane >> 4;             // A-frag: upper-K half lanes
    const int row0  = lane & 15;

    v8f c0 = {}; v8f c1 = {};
    const _Float16* wbTile = wb + (size_t)nt * KB_ * 32 * 16;

    const int kb0 = wave * (KB_ / 4), kb1 = kb0 + (KB_ / 4);
    for (int kb = kb0; kb < kb1; ++kb) {
        Frag16 a0, a1, b;
        // A: rows m = row0 (tile 0) / row0+16 (tile 1); lane half selects
        // K {0..7,16..23} vs {8..15,24..31} -> two contiguous 16 B loads each.
        const _Float16* aBase = rbuf + row0 * N_ + kb * 32 + khalf * 8;
        a0.q[0] = *(const uint4*)(aBase);
        a0.q[1] = *(const uint4*)(aBase + 16);
        a1.q[0] = *(const uint4*)(aBase + 16 * N_);
        a1.q[1] = *(const uint4*)(aBase + 16 * N_ + 16);
        // B: packed fragment, 32 B contiguous per lane.
        const _Float16* bBase = wbTile + ((size_t)kb * 32 + lane) * 16;
        b.q[0] = *(const uint4*)(bBase);
        b.q[1] = *(const uint4*)(bBase + 8);
        __builtin_prefetch(bBase + 512, 0, 0);   // next k-block of B
        c0 = __builtin_amdgcn_wmma_f32_16x16x32_f16(false, a0.v, false, b.v,
                                                    (short)0, c0, false, false);
        c1 = __builtin_amdgcn_wmma_f32_16x16x32_f16(false, a1.v, false, b.v,
                                                    (short)0, c1, false, false);
    }

    #pragma unroll
    for (int i = 0; i < 8; ++i) {
        part[((wave * 2 + 0) * 32 + lane) * 8 + i] = c0[i];
        part[((wave * 2 + 1) * 32 + lane) * 8 + i] = c1[i];
    }
    __syncthreads();

    if (wave < 2) {
        const int f    = wave;                      // row tile
        const int col  = nt * 16 + (lane & 15);     // C layout: N = lane&15
        const int mOff = f * 16 + ((lane >> 4) << 3); // lanes 16-31: M += 8
        const float dt = dtv[col];
        #pragma unroll
        for (int i = 0; i < 8; ++i) {
            float acc = 0.f;
            #pragma unroll
            for (int w = 0; w < 4; ++w)
                acc += part[((w * 2 + f) * 32 + lane) * 8 + i];
            const int m = mOff + i;
            const size_t gi = (size_t)m * N_ + col;
            const size_t ti = (size_t)t * B_ * N_ + gi;
            const float drive = acc + MUE_ + ext[ti];
            const float un = uPrev[gi] * (1.f - dt) + drive * dt;
            usave[ti]  = un;                          // also next step's u
            rnext[gi]  = (_Float16)sigmoid_act(un);   // next step's r
        }
    }
}

extern "C" void kernel_launch(void* const* d_in, const int* in_sizes, int n_in,
                              void* d_out, int out_size, void* d_ws, size_t ws_size,
                              hipStream_t stream) {
    const float* u0   = (const float*)d_in[0];
    const float* ext  = (const float*)d_in[1];
    const float* wrec = (const float*)d_in[2];
    const float* mask = (const float*)d_in[3];
    const float* sign = (const float*)d_in[4];
    const float* wout = (const float*)d_in[5];
    const float* dtv  = (const float*)d_in[6];

    float* usave   = (float*)d_out;                       // [500][32][2048]
    float* outsave = usave + (size_t)T_ * B_ * N_;        // [500][32][2]

    _Float16* rA = (_Float16*)d_ws;          // r ping   (128 KB)
    _Float16* rB = rA + B_ * N_;             // r pong   (128 KB)
    _Float16* wb = rB + B_ * N_;             // packed Wdale f16 (8 MB)

    init_r_kernel<<<dim3((B_ * N_ + 255) / 256), dim3(256), 0, stream>>>(u0, rA);
    prep_w_kernel<<<dim3((N_ * N_ + 255) / 256), dim3(256), 0, stream>>>(
        wrec, mask, sign, wb);

    const float* uPrev = u0;
    _Float16* rcur = rA; _Float16* rnxt = rB;
    for (int t = 0; t < T_; ++t) {
        readout_kernel<<<dim3(B_ * NOUT_), dim3(256), 0, stream>>>(
            rcur, wout, outsave, t);
        step_gemm_kernel<<<dim3(NT_), dim3(128), 0, stream>>>(
            rcur, wb, uPrev, ext, dtv, usave, rnxt, t);
        uPrev = usave + (size_t)t * B_ * N_;   // u_{t+1} lives in usave[t]
        _Float16* tmp = rcur; rcur = rnxt; rnxt = tmp;
    }
}
